// SelfAttentionHead_30485677867687
// MI455X (gfx1250) — matrix operand order repaired and checked by reading, not compiled, not run on hardware
//
#include <hip/hip_runtime.h>

// ---------------------------------------------------------------------------
// Problem constants: B=8, N=2048, M=256, KEY=64, D=128
// ---------------------------------------------------------------------------
#define BATCH    8
#define SEQ      2048
#define MDIM     256
#define KEY      64
#define DDIM     128
#define ROWS     (BATCH * SEQ)        // 16384

// 1/sqrt(KEY) * log2(e): folded into q so softmax uses raw exp2
#define QSCALE   (0.125f * 1.44269504088896f)

typedef __attribute__((ext_vector_type(16))) __bf16 v16bf;
typedef __attribute__((ext_vector_type(8)))  __bf16 v8bf;
typedef __attribute__((ext_vector_type(8)))  float  v8f;
typedef __attribute__((ext_vector_type(4)))  float  v4f;

static __device__ __forceinline__ float fast_exp2(float x) {
  return __builtin_amdgcn_exp2f(x);             // v_exp_f32 (base-2 native)
}

static __device__ __forceinline__ v16bf cat8(v8bf lo, v8bf hi) {
  return __builtin_shufflevector(lo, hi, 0, 1, 2, 3, 4, 5, 6, 7,
                                         8, 9, 10, 11, 12, 13, 14, 15);
}

// A fragment (16x32 bf16): per-lane two contiguous 16B runs.
static __device__ __forceinline__ v16bf load_a(const __bf16* src, int lda,
                                               int r, int half, int koff) {
  const __bf16* p = src + (size_t)r * lda + koff + half * 8;
  return cat8(*(const v8bf*)p, *(const v8bf*)(p + 16));
}

// B fragment (32x16 bf16) from K-contiguous storage srcT[n][K]:
// per-lane two contiguous 16B runs (same k order as load_a).
static __device__ __forceinline__ v16bf load_bT(const __bf16* srcT, int ldk,
                                                int n, int half, int koff) {
  const __bf16* p = srcT + (size_t)n * ldk + koff + half * 8;
  return cat8(*(const v8bf*)p, *(const v8bf*)(p + 16));
}

static __device__ __forceinline__ v8f wmma_bf16(v16bf a, v16bf b, v8f c) {
  return __builtin_amdgcn_wmma_f32_16x16x32_bf16(false, a, false, b,
                                                 (short)0, c, false, false);
}

// ---------------------------------------------------------------------------
// Kernel 0: convert f32 weight [R][C] -> transposed bf16 [C][R] (K-contiguous)
// ---------------------------------------------------------------------------
__global__ void cvtT_kernel(const float* __restrict__ src,
                            __bf16* __restrict__ dst, int R, int C) {
  int idx = blockIdx.x * 256 + threadIdx.x;
  if (idx >= R * C) return;
  int rr = idx / C, cc = idx - rr * C;
  dst[(size_t)cc * R + rr] = (__bf16)src[idx];
}

// ---------------------------------------------------------------------------
// Kernel 1: QKV projection.  x[16384,256]f32 @ WT[64,256]bf16 (x3)
// q is pre-scaled by 0.125*log2(e); v stored transposed [B][64][SEQ]
// ---------------------------------------------------------------------------
__global__ __launch_bounds__(128)
void qkv_kernel(const float* __restrict__ x,
                const __bf16* __restrict__ wqT,
                const __bf16* __restrict__ wkT,
                const __bf16* __restrict__ wvT,
                __bf16* __restrict__ q,
                __bf16* __restrict__ k,
                __bf16* __restrict__ vT) {
  const int lane = threadIdx.x & 31;
  const int wave = threadIdx.x >> 5;
  const int row0 = (blockIdx.x * 4 + wave) * 16;
  const int r    = lane & 15;
  const int half = lane >> 4;

  v8f accq[4] = {}, acck[4] = {}, accv[4] = {};

  for (int kk = 0; kk < MDIM; kk += 32) {
    // A fragment from fp32 x: two 32B runs, packed cvt to bf16
    const float* p = x + (size_t)(row0 + r) * MDIM + kk + half * 8;
    const v4f f0 = *(const v4f*)(p);
    const v4f f1 = *(const v4f*)(p + 4);
    const v4f f2 = *(const v4f*)(p + 16);
    const v4f f3 = *(const v4f*)(p + 20);
    v16bf a;
#pragma unroll
    for (int j = 0; j < 4; ++j) {
      a[j]      = (__bf16)f0[j];
      a[4 + j]  = (__bf16)f1[j];
      a[8 + j]  = (__bf16)f2[j];
      a[12 + j] = (__bf16)f3[j];
    }
#pragma unroll
    for (int nt = 0; nt < 4; ++nt) {
      const int col = nt * 16 + r;
      v16bf bq = load_bT(wqT, MDIM, col, half, kk);
      v16bf bk = load_bT(wkT, MDIM, col, half, kk);
      v16bf bv = load_bT(wvT, MDIM, col, half, kk);
      accq[nt] = wmma_bf16(a, bq, accq[nt]);
      acck[nt] = wmma_bf16(a, bk, acck[nt]);
      accv[nt] = wmma_bf16(a, bv, accv[nt]);
    }
  }
  // C layout: vgpr i -> row i + 8*half, col nt*16 + r
#pragma unroll
  for (int nt = 0; nt < 4; ++nt)
#pragma unroll
    for (int i = 0; i < 8; ++i) {
      const int gr  = row0 + half * 8 + i;
      const int col = nt * 16 + r;
      q[(size_t)gr * KEY + col] = (__bf16)(accq[nt][i] * QSCALE);
      k[(size_t)gr * KEY + col] = (__bf16)acck[nt][i];
      const int b  = gr >> 11;            // / SEQ
      const int sr = gr & (SEQ - 1);
      vT[((size_t)b * KEY + col) * SEQ + sr] = (__bf16)accv[nt][i];
    }
}

// ---------------------------------------------------------------------------
// Kernel 2: flash attention (exp2-domain online softmax).
// One wave per (batch, 16-row q tile).
// ---------------------------------------------------------------------------
__global__ __launch_bounds__(128)
void attn_kernel(const __bf16* __restrict__ q,
                 const __bf16* __restrict__ k,
                 const __bf16* __restrict__ vT,
                 __bf16* __restrict__ z) {
  __shared__ __bf16 plds[4][16 * 32];
  const int lane = threadIdx.x & 31;
  const int wave = threadIdx.x >> 5;
  const int w    = blockIdx.x * 4 + wave;       // 1024 waves
  const int b    = w >> 7;
  const int row0 = (w & 127) * 16;
  const int r    = lane & 15;
  const int half = lane >> 4;
  __bf16* pw = plds[wave];

  const __bf16* qb = q  + (size_t)b * SEQ * KEY;
  const __bf16* kb = k  + (size_t)b * SEQ * KEY;
  const __bf16* vb = vT + (size_t)b * KEY * SEQ;

  v16bf qa[2];
#pragma unroll
  for (int c = 0; c < 2; ++c)
    qa[c] = load_a(qb + (size_t)row0 * KEY, KEY, r, half, c * 32);

  v8f zacc[4] = {};
  float mrow[8], srow[8];
#pragma unroll
  for (int i = 0; i < 8; ++i) { mrow[i] = -1e30f; srow[i] = 0.0f; }

  for (int j0 = 0; j0 < SEQ; j0 += 32) {
    if (j0 + 32 < SEQ) {
      __builtin_prefetch(kb + (size_t)(j0 + 32) * KEY, 0, 3);
      __builtin_prefetch(vb + (size_t)(j0 + 32), 0, 3);
    }
    // ---- scores (already in log2 domain via pre-scaled q) ----
    v8f s[2];
#pragma unroll
    for (int nt = 0; nt < 2; ++nt) {
      v8f acc = {};
#pragma unroll
      for (int c = 0; c < 2; ++c) {
        v16bf bkf = load_bT(kb, KEY, j0 + nt * 16 + r, half, c * 32);
        acc = wmma_bf16(qa[c], bkf, acc);
      }
      s[nt] = acc;
    }
    // ---- online softmax, base-2 ----
    float newm[8];
#pragma unroll
    for (int i = 0; i < 8; ++i) {
      float mx = fmaxf(s[0][i], s[1][i]);
#pragma unroll
      for (int msk = 8; msk >= 1; msk >>= 1)
        mx = fmaxf(mx, __shfl_xor(mx, msk, 32));
      newm[i] = fmaxf(mrow[i], mx);
    }
#pragma unroll
    for (int i = 0; i < 8; ++i) {
      float e0 = fast_exp2(s[0][i] - newm[i]);
      float e1 = fast_exp2(s[1][i] - newm[i]);
      s[0][i] = e0; s[1][i] = e1;
      float rs = e0 + e1;
#pragma unroll
      for (int msk = 8; msk >= 1; msk >>= 1)
        rs += __shfl_xor(rs, msk, 32);
      float corr = fast_exp2(mrow[i] - newm[i]);
      srow[i] = srow[i] * corr + rs;
      mrow[i] = newm[i];
#pragma unroll
      for (int nt = 0; nt < 4; ++nt) zacc[nt][i] = zacc[nt][i] * corr;
    }
    // ---- C-layout -> A-layout via per-wave LDS bounce ----
#pragma unroll
    for (int nt = 0; nt < 2; ++nt)
#pragma unroll
      for (int i = 0; i < 8; ++i)
        pw[(half * 8 + i) * 32 + nt * 16 + r] = (__bf16)s[nt][i];
    asm volatile("s_wait_dscnt 0" ::: "memory");
    v16bf pa = load_a(pw, 32, r, half, 0);
    // ---- z += P(16x32) @ V(32x64), V read K-contiguous from vT ----
#pragma unroll
    for (int dt = 0; dt < 4; ++dt) {
      v16bf bvf = load_bT(vb, SEQ, dt * 16 + r, half, j0);
      zacc[dt] = wmma_bf16(pa, bvf, zacc[dt]);
    }
  }
  // ---- finalize ----
#pragma unroll
  for (int dt = 0; dt < 4; ++dt)
#pragma unroll
    for (int i = 0; i < 8; ++i) {
      const size_t idx = (size_t)(b * SEQ + row0 + half * 8 + i) * KEY + dt * 16 + r;
      z[idx] = (__bf16)(zacc[dt][i] / srow[i]);
    }
}

// ---------------------------------------------------------------------------
// Kernel 3: 3-layer ELU MLP.  z[16x64] -> 128 -> 128 -> 128 (fp32 out).
// ---------------------------------------------------------------------------
__global__ __launch_bounds__(128)
void mlp_kernel(const __bf16* __restrict__ z,
                const __bf16* __restrict__ w1T, const float* __restrict__ b1,
                const __bf16* __restrict__ w2T, const float* __restrict__ b2,
                const __bf16* __restrict__ w3T, const float* __restrict__ b3,
                float* __restrict__ out) {
  __shared__ __bf16 hlds[4][16 * DDIM];
  const int lane = threadIdx.x & 31;
  const int wave = threadIdx.x >> 5;
  const int row0 = (blockIdx.x * 4 + wave) * 16;
  const int r    = lane & 15;
  const int half = lane >> 4;
  __bf16* hw = hlds[wave];

  // ---- layer 1: z @ W1 + b1, ELU ----
  v16bf za[2];
#pragma unroll
  for (int c = 0; c < 2; ++c)
    za[c] = load_a(z + (size_t)row0 * KEY, KEY, r, half, c * 32);

  v8f h[8];
#pragma unroll
  for (int ct = 0; ct < 8; ++ct) {
    const float bias = b1[ct * 16 + r];
    v8f acc;
#pragma unroll
    for (int i = 0; i < 8; ++i) acc[i] = bias;
#pragma unroll
    for (int c = 0; c < 2; ++c) {
      v16bf bw = load_bT(w1T, KEY, ct * 16 + r, half, c * 32);
      acc = wmma_bf16(za[c], bw, acc);
    }
#pragma unroll
    for (int i = 0; i < 8; ++i) {
      float x0 = acc[i];
      acc[i] = x0 > 0.0f ? x0 : (__expf(x0) - 1.0f);
    }
    h[ct] = acc;
  }

  // ---- layers 2 and 3: LDS bounce then K=128 matmul ----
  const __bf16* Ws[2] = { w2T, w3T };
  const float*  bs[2] = { b2, b3 };
#pragma unroll 1
  for (int layer = 0; layer < 2; ++layer) {
#pragma unroll
    for (int ct = 0; ct < 8; ++ct)
#pragma unroll
      for (int i = 0; i < 8; ++i)
        hw[(half * 8 + i) * DDIM + ct * 16 + r] = (__bf16)h[ct][i];
    asm volatile("s_wait_dscnt 0" ::: "memory");
    v16bf ha[4];
#pragma unroll
    for (int c = 0; c < 4; ++c)
      ha[c] = load_a(hw, DDIM, r, half, c * 32);
    const __bf16* W  = Ws[layer];
    const float*  bb = bs[layer];
#pragma unroll
    for (int ct = 0; ct < 8; ++ct) {
      const float bias = bb[ct * 16 + r];
      v8f acc;
#pragma unroll
      for (int i = 0; i < 8; ++i) acc[i] = bias;
#pragma unroll
      for (int c = 0; c < 4; ++c) {
        v16bf bw = load_bT(W, DDIM, ct * 16 + r, half, c * 32);
        acc = wmma_bf16(ha[c], bw, acc);
      }
#pragma unroll
      for (int i = 0; i < 8; ++i) {
        float x0 = acc[i];
        acc[i] = x0 > 0.0f ? x0 : (__expf(x0) - 1.0f);
      }
      h[ct] = acc;
    }
  }

  // ---- fp32 output [ROWS][128] ----
#pragma unroll
  for (int ct = 0; ct < 8; ++ct)
#pragma unroll
    for (int i = 0; i < 8; ++i)
      out[(size_t)(row0 + half * 8 + i) * DDIM + ct * 16 + r] = h[ct][i];
}

// ---------------------------------------------------------------------------
// Launcher
// ---------------------------------------------------------------------------
extern "C" void kernel_launch(void* const* d_in, const int* in_sizes, int n_in,
                              void* d_out, int out_size, void* d_ws, size_t ws_size,
                              hipStream_t stream) {
  (void)in_sizes; (void)n_in; (void)out_size; (void)ws_size;
  const float* x  = (const float*)d_in[0];
  const float* Wq = (const float*)d_in[1];
  const float* Wk = (const float*)d_in[2];
  const float* Wv = (const float*)d_in[3];
  const float* W1 = (const float*)d_in[4];
  const float* b1 = (const float*)d_in[5];
  const float* W2 = (const float*)d_in[6];
  const float* b2 = (const float*)d_in[7];
  const float* W3 = (const float*)d_in[8];
  const float* b3 = (const float*)d_in[9];

  __bf16* q   = (__bf16*)d_ws;                  // [ROWS][64]
  __bf16* k   = q   + (size_t)ROWS * KEY;       // [ROWS][64]
  __bf16* vT  = k   + (size_t)ROWS * KEY;       // [B][64][SEQ]
  __bf16* z   = vT  + (size_t)ROWS * KEY;       // [ROWS][64]
  __bf16* wqT = z   + (size_t)ROWS * KEY;       // [64][256]
  __bf16* wkT = wqT + (size_t)KEY * MDIM;
  __bf16* wvT = wkT + (size_t)KEY * MDIM;
  __bf16* w1T = wvT + (size_t)KEY * MDIM;       // [128][64]
  __bf16* w2T = w1T + (size_t)DDIM * KEY;       // [128][128]
  __bf16* w3T = w2T + (size_t)DDIM * DDIM;      // [128][128]

  // Prologue: one-time weight convert + transpose (K-contiguous bf16)
  cvtT_kernel<<<(MDIM * KEY + 255) / 256, 256, 0, stream>>>(Wq, wqT, MDIM, KEY);
  cvtT_kernel<<<(MDIM * KEY + 255) / 256, 256, 0, stream>>>(Wk, wkT, MDIM, KEY);
  cvtT_kernel<<<(MDIM * KEY + 255) / 256, 256, 0, stream>>>(Wv, wvT, MDIM, KEY);
  cvtT_kernel<<<(KEY * DDIM + 255) / 256, 256, 0, stream>>>(W1, w1T, KEY, DDIM);
  cvtT_kernel<<<(DDIM * DDIM + 255) / 256, 256, 0, stream>>>(W2, w2T, DDIM, DDIM);
  cvtT_kernel<<<(DDIM * DDIM + 255) / 256, 256, 0, stream>>>(W3, w3T, DDIM, DDIM);

  qkv_kernel <<<ROWS / 16 / 4, 128, 0, stream>>>(x, wqT, wkT, wvT, q, k, vT);
  attn_kernel<<<ROWS / 16 / 4, 128, 0, stream>>>(q, k, vT, z);
  mlp_kernel <<<ROWS / 16 / 4, 128, 0, stream>>>(z, w1T, b1, w2T, b2, w3T, b3,
                                                 (float*)d_out);
}
